// MultiHeadLocationAwareAttention_14551349199360
// MI455X (gfx1250) — matrix operand order repaired and checked by reading, not compiled, not run on hardware
//
#include <hip/hip_runtime.h>
#include <hip/hip_bf16.h>
#include <cstdint>

// Problem dims (fixed by the reference)
#define BB   16
#define SS   4096
#define HIDD 1024
#define NHH  16
#define CO   10
#define DIMD 64
#define TK   32      // WMMA K step (bf16 16x16x32)
#define BM   128     // workgroup tile M
#define BN   64      // workgroup tile N
#define LSTR 40      // LDS row stride in halves (80B: 16B-aligned, conflict-free)

typedef __attribute__((ext_vector_type(16))) __bf16 v16bf;
typedef __attribute__((ext_vector_type(8)))  __bf16 v8bf;
typedef __attribute__((ext_vector_type(8)))  float  v8f;

// two 16B LDS loads -> one 32B fragment (lands in consecutive VGPRs)
__device__ __forceinline__ v16bf lds_ld16(const __bf16* p) {
    v8bf lo = *(const v8bf*)p;
    v8bf hi = *(const v8bf*)(p + 8);
    return __builtin_shufflevector(lo, hi, 0, 1, 2, 3, 4, 5, 6, 7,
                                           8, 9, 10, 11, 12, 13, 14, 15);
}

// ---------------------------------------------------------------------------
// K0: q = query @ W_q   -> (B, NH*DIM)   [tiny, VALU]
// ---------------------------------------------------------------------------
__global__ void qproj_kernel(const float* __restrict__ query,
                             const float* __restrict__ Wq,
                             float* __restrict__ qout) {
    const int b = blockIdx.x;
    const int o = threadIdx.x;              // 0..1023
    const float* qr = query + b * HIDD;
    float acc = 0.f;
    #pragma unroll 8
    for (int k = 0; k < HIDD; ++k)
        acc = fmaf(qr[k], Wq[k * HIDD + o], acc);
    qout[b * HIDD + o] = acc;
}

// ---------------------------------------------------------------------------
// K1: conv1d(last_attn, pad=1) + bias, then loc = tanh(conv . W_loc)
// ---------------------------------------------------------------------------
__global__ void conv_loc_kernel(const float* __restrict__ la,
                                const float* __restrict__ cw,
                                const float* __restrict__ cb,
                                const float* __restrict__ Wl,
                                float* __restrict__ loc) {
    const int b = blockIdx.y;
    const int s = blockIdx.x * 64 + threadIdx.x;
    float conv[CO];
    #pragma unroll
    for (int c = 0; c < CO; ++c) conv[c] = cb[c];
    for (int h = 0; h < NHH; ++h) {
        const float* row = la + ((size_t)b * NHH + h) * SS;
        const float xm = (s > 0)      ? row[s - 1] : 0.f;
        const float x0 =                row[s];
        const float xp = (s < SS - 1) ? row[s + 1] : 0.f;
        #pragma unroll
        for (int c = 0; c < CO; ++c) {
            const float* w = cw + ((size_t)c * NHH + h) * 3;
            conv[c] = fmaf(xm, w[0], fmaf(x0, w[1], fmaf(xp, w[2], conv[c])));
        }
    }
    float* out = loc + ((size_t)b * SS + s) * DIMD;
    for (int d = 0; d < DIMD; ++d) {
        float acc = 0.f;
        #pragma unroll
        for (int c = 0; c < CO; ++c)
            acc = fmaf(conv[c], Wl[c * DIMD + d], acc);
        out[d] = tanhf(acc);
    }
}

// ---------------------------------------------------------------------------
// K2: vproj = value(M=B*S x 1024) @ W_v(1024 x 1024), bf16 WMMA, bf16 out.
// 128x64 macro-tile, 8 wave32 waves, wave = 16-row strip x 4 N-tiles
// (4 WMMAs / wave / K-step). Double-buffered fragment-ready LDS:
//   A: Asp[row][hi][slot]  (ISA 16-bit A lane order, 16 contiguous halves/lane)
//   B: BsT[col][hi][slot]  (per column, same k order as A, 16 contiguous halves/lane)
// grid: (M/128, 1024/64) = (512, 16), 256 threads
// ---------------------------------------------------------------------------
__global__ void __launch_bounds__(256)
vproj_wmma_kernel(const float* __restrict__ A,
                  const float* __restrict__ Bw,
                  __bf16* __restrict__ Dst) {
    __shared__ __align__(16) __bf16 Asp[2][BM * LSTR];
    __shared__ __align__(16) __bf16 BsT[2][BN * LSTR];

    const int m0   = blockIdx.x * BM;
    const int n0   = blockIdx.y * BN;
    const int tid  = threadIdx.x;
    const int lane = tid & 31;
    const int wave = tid >> 5;           // tile row strip 0..7
    const int lr   = lane & 15;
    const int hi   = lane >> 4;          // lane half (ISA 16-bit layouts)

    v8f acc[4] = {{}, {}, {}, {}};

    // --- staging helpers -------------------------------------------------
    // A: thread covers 16 consecutive K of one row (two 8-chunks).
    const int ar = tid >> 1;             // 0..127
    const int ac = (tid & 1) * 16;       // 0 or 16
    // B: thread covers 8 consecutive K of one column.
    const int bcol = tid & 63;
    const int bkc  = (tid >> 6) * 8;     // 0,8,16,24

    auto stageA = [&](int p, int k0) {
        const float* src = A + (size_t)(m0 + ar) * HIDD + k0 + ac;
        #pragma unroll
        for (int ch = 0; ch < 2; ++ch) {
            const int c     = ac + ch * 8;         // global k offset in tile
            const int chi   = (c >> 3) & 1;        // lane-half this chunk feeds
            const int slot0 = (c & 16) >> 1;       // 0 or 8
            v8bf t;
            #pragma unroll
            for (int j = 0; j < 8; ++j) t[j] = (__bf16)src[ch * 8 + j];
            *(v8bf*)&Asp[p][ar * LSTR + chi * 16 + slot0] = t;
        }
    };
    auto stageB = [&](int p, int k0) {
        v8bf t;
        #pragma unroll
        for (int j = 0; j < 8; ++j)
            t[j] = (__bf16)Bw[(size_t)(k0 + bkc + j) * HIDD + n0 + bcol];
        *(v8bf*)&BsT[p][bcol * LSTR + ((bkc >> 3) & 1) * 16 + ((bkc & 16) >> 1)] = t;
    };

    stageA(0, 0);
    stageB(0, 0);

    const int NT = HIDD / TK;            // 32 K-steps
    for (int kt = 0; kt < NT; ++kt) {
        __syncthreads();
        const int p = kt & 1;
        if (kt + 1 < NT) {
            __builtin_prefetch(A + (size_t)(m0 + ar) * HIDD + (kt + 1) * TK + ac, 0, 1);
            stageA(p ^ 1, (kt + 1) * TK);
            stageB(p ^ 1, (kt + 1) * TK);
        }
        // fragment loads: 32 contiguous bytes per lane (2x ds_load_b128)
        const v16bf af = lds_ld16(&Asp[p][(wave * 16 + lr) * LSTR + hi * 16]);
        #pragma unroll
        for (int t = 0; t < 4; ++t) {
            const v16bf bf = lds_ld16(&BsT[p][(t * 16 + lr) * LSTR + hi * 16]);
            acc[t] = __builtin_amdgcn_wmma_f32_16x16x32_bf16(
                         false, af, false, bf, (short)0, acc[t], false, false);
        }
    }

    // D layout (16x16 f32): n = lane&15, m = i + 8*(lane>>4)
    const int mb = m0 + wave * 16 + hi * 8;
    #pragma unroll
    for (int t = 0; t < 4; ++t) {
        const int n = n0 + t * 16 + lr;
        #pragma unroll
        for (int i = 0; i < 8; ++i)
            Dst[(size_t)(mb + i) * HIDD + n] = (__bf16)acc[t][i];
    }
}

// ---------------------------------------------------------------------------
// K3: score[n,s] = b_score + sum_d w_score[d]*tanh(vproj[b,s,h*64+d]+q+loc+bias)
// ---------------------------------------------------------------------------
__global__ void score_kernel(const __bf16* __restrict__ vproj,
                             const float* __restrict__ qout,
                             const float* __restrict__ loc,
                             const float* __restrict__ bias,
                             const float* __restrict__ wsc,
                             const float* __restrict__ bsc,
                             float* __restrict__ score) {
    const int tid = blockIdx.x * blockDim.x + threadIdx.x;
    const int s = tid & (SS - 1);
    const int n = tid >> 12;
    const int b = n >> 4;
    const int h = n & 15;
    const __bf16* vp = vproj + ((size_t)b * SS + s) * HIDD + h * DIMD;
    const float*  lc = loc   + ((size_t)b * SS + s) * DIMD;
    const float*  qh = qout  + ((size_t)b * NHH + h) * DIMD;
    float acc = bsc[0];
    #pragma unroll 8
    for (int d = 0; d < DIMD; ++d) {
        const float t = tanhf((float)vp[d] + qh[d] + lc[d] + bias[d]);
        acc = fmaf(t, wsc[d], acc);
    }
    score[(size_t)n * SS + s] = acc;
}

// ---------------------------------------------------------------------------
// K4: softmax over s per row n. grid: B*NH blocks, 256 threads
// ---------------------------------------------------------------------------
__global__ void softmax_kernel(const float* __restrict__ score,
                               float* __restrict__ attn) {
    __shared__ float red[256];
    const int n = blockIdx.x;
    const int t = threadIdx.x;
    const float* row = score + (size_t)n * SS;
    float m = -INFINITY;
    for (int s = t; s < SS; s += 256) m = fmaxf(m, row[s]);
    red[t] = m; __syncthreads();
    for (int o = 128; o > 0; o >>= 1) {
        if (t < o) red[t] = fmaxf(red[t], red[t + o]);
        __syncthreads();
    }
    m = red[0]; __syncthreads();
    float sum = 0.f;
    for (int s = t; s < SS; s += 256) sum += expf(row[s] - m);
    red[t] = sum; __syncthreads();
    for (int o = 128; o > 0; o >>= 1) {
        if (t < o) red[t] += red[t + o];
        __syncthreads();
    }
    const float inv = 1.f / red[0];
    for (int s = t; s < SS; s += 256)
        attn[(size_t)n * SS + s] = expf(row[s] - m) * inv;
}

// ---------------------------------------------------------------------------
// K5: context with the reference's buffer-reinterpreting v2 view:
// context[b,h2,d2] = sum_s2 attn[n,s2] * vproj[b, (s2&255)*16+h2, (s2>>8)*64+d2]
// ---------------------------------------------------------------------------
__global__ void context_kernel(const __bf16* __restrict__ vproj,
                               const float* __restrict__ attn,
                               float* __restrict__ ctx) {
    const int n  = blockIdx.x;
    const int b  = n >> 4;
    const int h2 = n & 15;
    const int d2 = threadIdx.x;
    float acc = 0.f;
    for (int s2 = 0; s2 < SS; ++s2) {
        const int si  = ((s2 & 255) << 4) + h2;
        const int col = ((s2 >> 8) << 6) + d2;
        acc = fmaf(attn[(size_t)n * SS + s2],
                   (float)vproj[((size_t)b * SS + si) * HIDD + col], acc);
    }
    ctx[b * HIDD + h2 * DIMD + d2] = acc;
}

// ---------------------------------------------------------------------------
extern "C" void kernel_launch(void* const* d_in, const int* in_sizes, int n_in,
                              void* d_out, int out_size, void* d_ws, size_t ws_size,
                              hipStream_t stream) {
    const float* query   = (const float*)d_in[0];   // (B,1,HID)
    const float* value   = (const float*)d_in[1];   // (B,S,HID)
    const float* lastatt = (const float*)d_in[2];   // (B,NH,S)
    const float* conv_w  = (const float*)d_in[3];   // (COUT,NH,3)
    const float* conv_b  = (const float*)d_in[4];   // (COUT,)
    const float* W_loc   = (const float*)d_in[5];   // (COUT,DIM)
    const float* W_q     = (const float*)d_in[6];   // (HID,HID)
    const float* W_v     = (const float*)d_in[7];   // (HID,HID)
    const float* w_score = (const float*)d_in[8];   // (DIM,1)
    const float* b_score = (const float*)d_in[9];   // (1,)
    const float* bias    = (const float*)d_in[10];  // (DIM,)

    float* out_ctx  = (float*)d_out;                         // (B,1,NH*DIM)
    float* out_attn = out_ctx + (size_t)BB * NHH * DIMD;     // (B,NH,S)

    // workspace layout
    float* q_ws     = (float*)d_ws;                                  // B*HID
    float* loc_ws   = q_ws   + (size_t)BB * HIDD;                    // B*S*DIM
    float* score_ws = loc_ws + (size_t)BB * SS * DIMD;               // B*NH*S
    __bf16* vproj   = (__bf16*)(score_ws + (size_t)BB * NHH * SS);   // B*S*HID bf16

    qproj_kernel<<<BB, 1024, 0, stream>>>(query, W_q, q_ws);
    conv_loc_kernel<<<dim3(SS / 64, BB), 64, 0, stream>>>(lastatt, conv_w, conv_b,
                                                          W_loc, loc_ws);
    vproj_wmma_kernel<<<dim3((BB * SS) / BM, HIDD / BN), 256, 0, stream>>>(
        value, W_v, vproj);
    score_kernel<<<(BB * NHH * SS) / 256, 256, 0, stream>>>(
        vproj, q_ws, loc_ws, bias, w_score, b_score, score_ws);
    softmax_kernel<<<BB * NHH, 256, 0, stream>>>(score_ws, out_attn);
    context_kernel<<<BB * NHH, 64, 0, stream>>>(vproj, out_attn, out_ctx);
}
